// PocketGNN_58291296141578
// MI455X (gfx1250) — compile-verified, matrix operand order checked by reading, not run on hardware
//
#include <hip/hip_runtime.h>
#include <hip/hip_bf16.h>

#define NNODES   16384
#define NEDGES   262144
#define NDIM     480
#define HID      256
#define OUTD     128
#define NGRAPHS  256
#define E2TOT    (NEDGES + NNODES)

typedef __attribute__((ext_vector_type(16))) __bf16          v16bf;
typedef __attribute__((ext_vector_type(8)))  float           v8f;
typedef __attribute__((ext_vector_type(8)))  unsigned short  v8us;
typedef __attribute__((ext_vector_type(16))) unsigned short  v16us;

// ---------- helpers ----------
__device__ __forceinline__ unsigned f2key(float f) {
  unsigned u = __float_as_uint(f);
  return (u & 0x80000000u) ? ~u : (u | 0x80000000u);
}
__device__ __forceinline__ float key2f(unsigned k) {
  return (k & 0x80000000u) ? __uint_as_float(k ^ 0x80000000u)
                           : __uint_as_float(~k);
}

// A fragment: lane holds row (lane&15); K = kb..kb+7 (16B) and kb+16..kb+23 (16B)
__device__ __forceinline__ v16bf load_frag_a(const __bf16* p) {
  v8us lo = *(const v8us*)p;
  v8us hi = *(const v8us*)(p + 16);
  v16us c = __builtin_shufflevector(lo, hi, 0, 1, 2, 3, 4, 5, 6, 7,
                                    8, 9, 10, 11, 12, 13, 14, 15);
  return __builtin_bit_cast(v16bf, c);
}

// ---------- fp32 -> bf16 convert ----------
__global__ void cvt_bf16_kernel(const float* __restrict__ in,
                                __bf16* __restrict__ out, int n) {
  int t = blockIdx.x * blockDim.x + threadIdx.x;
  if (t < n) out[t] = (__bf16)in[t];
}

// ---------- pack weight B[K,N] (f32) into per-lane WMMA fragments (bf16) ----
// packB[((kt*ntiles + nt)*32 + lane)*16 + i] = B[kt*32 + kmap(lane,i)][nt*16 + (lane&15)]
__global__ void pack_b_kernel(const float* __restrict__ B,
                              __bf16* __restrict__ packB, int K, int N) {
  int t = blockIdx.x * blockDim.x + threadIdx.x;
  if (t >= K * N) return;
  int i    = t & 15;
  int lane = (t >> 4) & 31;
  int tile = t >> 9;
  int ntiles = N >> 4;
  int kt = tile / ntiles, nt = tile - kt * ntiles;
  int kk = kt * 32 + ((lane >> 4) << 3) + (i < 8 ? i : i + 8);
  int nn = nt * 16 + (lane & 15);
  packB[t] = (__bf16)B[(size_t)kk * N + nn];
}

// ---------- WMMA bf16 GEMM: C[M,N] = A[M,K] @ B[K,N] (+bias) ----------
// One wave computes a 32x64 tile: 2 A fragments x 4 packed B fragments.
__global__ __launch_bounds__(32)
void gemm_bf16_wmma(const __bf16* __restrict__ A, const __bf16* __restrict__ packB,
                    const float* __restrict__ bias, float* __restrict__ C,
                    int M, int K, int N) {
  const int lane  = threadIdx.x & 31;
  const int tn    = blockIdx.x;                 // N/64 tiles
  const int tm    = blockIdx.y;                 // M/32 tiles
  const int l15   = lane & 15;
  const int khalf = (lane >> 4) << 3;           // 0 or 8
  const int ntiles = N >> 4;
  const int row0  = tm * 32 + l15;
  const int row1  = row0 + 16;

  v8f acc[2][4] = {};
  const int ksteps = K >> 5;
  for (int kt = 0; kt < ksteps; ++kt) {
    const __bf16* abase = A + (size_t)kt * 32 + khalf;
    v16bf a0 = load_frag_a(abase + (size_t)row0 * K);
    v16bf a1 = load_frag_a(abase + (size_t)row1 * K);
#pragma unroll
    for (int s = 0; s < 4; ++s) {
      const int ntile = tn * 4 + s;
      v16us bu = *(const v16us*)(packB + (((size_t)kt * ntiles + ntile) * 32 + lane) * 16);
      v16bf bv = __builtin_bit_cast(v16bf, bu);
      acc[0][s] = __builtin_amdgcn_wmma_f32_16x16x32_bf16(
          false, a0, false, bv, (short)0, acc[0][s], false, false);
      acc[1][s] = __builtin_amdgcn_wmma_f32_16x16x32_bf16(
          false, a1, false, bv, (short)0, acc[1][s], false, false);
    }
  }
  // D layout: VGPR v -> M = v + 8*(lane>=16), N = lane&15
  const int mBase = tm * 32 + ((lane >> 4) << 3);
#pragma unroll
  for (int s = 0; s < 4; ++s) {
    const int col = tn * 64 + s * 16 + l15;
    const float bb = bias ? bias[col] : 0.0f;
#pragma unroll
    for (int mi = 0; mi < 2; ++mi)
#pragma unroll
      for (int v = 0; v < 8; ++v)
        C[(size_t)(mBase + mi * 16 + v) * N + col] = acc[mi][s][v] + bb;
  }
}

// ---------- degree + edge-attr sum (self-loop fill_value='mean') ----------
__global__ void deg_easum_kernel(const int* __restrict__ dst,
                                 const float* __restrict__ ea,
                                 float* __restrict__ deg,
                                 float* __restrict__ easum, int E) {
  int t = blockIdx.x * blockDim.x + threadIdx.x;
  if (t < E) {
    int d = dst[t];
    atomicAdd(&deg[d], 1.0f);
    atomicAdd(&easum[d], ea[t]);
  }
}
__global__ void loopea_kernel(const float* __restrict__ easum,
                              const float* __restrict__ deg,
                              float* __restrict__ loop_ea, int n) {
  int t = blockIdx.x * blockDim.x + threadIdx.x;
  if (t < n) loop_ea[t] = easum[t] / fmaxf(deg[t], 1.0f);
}

// ---------- per-edge attention logits + segment max ----------
template <int DOUT, int H>
__global__ void edge_alpha_kernel(const int* __restrict__ src,
                                  const int* __restrict__ dst,
                                  const float* __restrict__ ea,
                                  const float* __restrict__ loop_ea,
                                  const float* __restrict__ xl,
                                  const float* __restrict__ xr,
                                  const float* __restrict__ We,
                                  const float* __restrict__ att,
                                  float* __restrict__ alpha,
                                  unsigned* __restrict__ amaxKey) {
  constexpr int CH = DOUT / 32;       // channels per lane
  constexpr int C  = DOUT / H;
  constexpr int G  = C / CH;          // lanes per head
  int wid  = (blockIdx.x * blockDim.x + threadIdx.x) >> 5;
  int lane = threadIdx.x & 31;
  if (wid >= E2TOT) return;
  int s, d; float w;
  if (wid < NEDGES) { s = src[wid]; d = dst[wid]; w = ea[wid]; }
  else              { s = d = wid - NEDGES; w = loop_ea[s]; }
  const int ch0 = lane * CH;
  const int h   = ch0 / C;
  const float4* xls  = (const float4*)(xl + (size_t)s * DOUT + ch0);
  const float4* xrd  = (const float4*)(xr + (size_t)d * DOUT + ch0);
  const float4* wep  = (const float4*)(We + ch0);
  const float4* attp = (const float4*)(att + ch0);
  float partial = 0.0f;
#pragma unroll
  for (int j = 0; j < CH / 4; ++j) {
    float4 a = xls[j], b = xrd[j], e = wep[j], t = attp[j];
    float m0 = a.x + b.x + w * e.x; m0 = (m0 > 0.f) ? m0 : 0.2f * m0;
    float m1 = a.y + b.y + w * e.y; m1 = (m1 > 0.f) ? m1 : 0.2f * m1;
    float m2 = a.z + b.z + w * e.z; m2 = (m2 > 0.f) ? m2 : 0.2f * m2;
    float m3 = a.w + b.w + w * e.w; m3 = (m3 > 0.f) ? m3 : 0.2f * m3;
    partial += m0 * t.x + m1 * t.y + m2 * t.z + m3 * t.w;
  }
#pragma unroll
  for (int off = G >> 1; off > 0; off >>= 1)
    partial += __shfl_xor(partial, off, 32);
  if ((lane & (G - 1)) == 0) {
    alpha[(size_t)wid * H + h] = partial;
    atomicMax(&amaxKey[d * H + h], f2key(partial));
  }
}

// ---------- exp(alpha - max) + denominator ----------
template <int H>
__global__ void edge_exp_kernel(const int* __restrict__ dst,
                                const float* __restrict__ alpha,
                                const unsigned* __restrict__ amaxKey,
                                float* __restrict__ ex,
                                float* __restrict__ denom) {
  int t = blockIdx.x * blockDim.x + threadIdx.x;
  if (t >= E2TOT * H) return;
  int e = t / H, h = t - e * H;
  int d = (e < NEDGES) ? dst[e] : (e - NEDGES);
  float v = expf(alpha[t] - key2f(amaxKey[d * H + h]));
  ex[t] = v;
  atomicAdd(&denom[d * H + h], v);
}

// ---------- weighted message scatter: out[dst] += a * xl[src] ----------
template <int DOUT, int H>
__global__ void edge_scatter_kernel(const int* __restrict__ src,
                                    const int* __restrict__ dst,
                                    const float* __restrict__ xl,
                                    const float* __restrict__ ex,
                                    const float* __restrict__ denom,
                                    float* __restrict__ out) {
  constexpr int CH = DOUT / 32;
  constexpr int C  = DOUT / H;
  int wid  = (blockIdx.x * blockDim.x + threadIdx.x) >> 5;
  int lane = threadIdx.x & 31;
  if (wid >= E2TOT) return;
  int s, d;
  if (wid < NEDGES) { s = src[wid]; d = dst[wid]; }
  else              { s = d = wid - NEDGES; }
  const int ch0 = lane * CH;
  const int h   = ch0 / C;
  float a = ex[(size_t)wid * H + h] / denom[d * H + h];
  const float4* xls = (const float4*)(xl + (size_t)s * DOUT + ch0);
  float* op = out + (size_t)d * DOUT + ch0;
#pragma unroll
  for (int j = 0; j < CH / 4; ++j) {
    float4 v = xls[j];
    atomicAdd(op + 4 * j + 0, a * v.x);
    atomicAdd(op + 4 * j + 1, a * v.y);
    atomicAdd(op + 4 * j + 2, a * v.z);
    atomicAdd(op + 4 * j + 3, a * v.w);
  }
}

// ---------- +bias, layernorm, (gelu), (+residual); optional bf16 copy ----------
template <int DOUT>
__global__ __launch_bounds__(32)
void node_post_kernel(const float* __restrict__ acc, const float* __restrict__ bo,
                      const float* __restrict__ g, const float* __restrict__ b,
                      const float* __restrict__ resAdd,
                      float* __restrict__ outF, __bf16* __restrict__ outB,
                      int doGelu) {
  constexpr int CH = DOUT / 32;
  int n = blockIdx.x, lane = threadIdx.x;
  const int ch0 = lane * CH;
  float vals[CH];
  const float4* ap = (const float4*)(acc + (size_t)n * DOUT + ch0);
  const float4* bp = (const float4*)(bo + ch0);
  float sum = 0.0f;
#pragma unroll
  for (int j = 0; j < CH / 4; ++j) {
    float4 a = ap[j], bb = bp[j];
    vals[4 * j + 0] = a.x + bb.x; vals[4 * j + 1] = a.y + bb.y;
    vals[4 * j + 2] = a.z + bb.z; vals[4 * j + 3] = a.w + bb.w;
    sum += vals[4 * j] + vals[4 * j + 1] + vals[4 * j + 2] + vals[4 * j + 3];
  }
#pragma unroll
  for (int off = 16; off > 0; off >>= 1) sum += __shfl_xor(sum, off, 32);
  float mu = sum / (float)DOUT;
  float var = 0.0f;
#pragma unroll
  for (int j = 0; j < CH; ++j) { float dd = vals[j] - mu; var += dd * dd; }
#pragma unroll
  for (int off = 16; off > 0; off >>= 1) var += __shfl_xor(var, off, 32);
  var /= (float)DOUT;
  float rs = rsqrtf(var + 1e-5f);
#pragma unroll
  for (int j = 0; j < CH; ++j) {
    int ch = ch0 + j;
    float v = (vals[j] - mu) * rs * g[ch] + b[ch];
    if (doGelu) v = 0.5f * v * (1.0f + erff(v * 0.70710678118654752f));
    if (resAdd) v += resAdd[(size_t)n * DOUT + ch];
    outF[(size_t)n * DOUT + ch] = v;
    if (outB) outB[(size_t)n * DOUT + ch] = (__bf16)v;
  }
}

// ---------- global mean pool ----------
__global__ void pool_scatter_kernel(const float* __restrict__ h3,
                                    const int* __restrict__ batch,
                                    float* __restrict__ sums,
                                    float* __restrict__ cnt) {
  int t = blockIdx.x * blockDim.x + threadIdx.x;
  if (t >= NNODES * OUTD) return;
  int n = t >> 7, c = t & (OUTD - 1);
  int gg = batch[n];
  atomicAdd(&sums[gg * OUTD + c], h3[t]);
  if (c == 0) atomicAdd(&cnt[gg], 1.0f);
}
__global__ void pool_div_kernel(const float* __restrict__ sums,
                                const float* __restrict__ cnt,
                                float* __restrict__ out) {
  int t = blockIdx.x * blockDim.x + threadIdx.x;
  if (t >= NGRAPHS * OUTD) return;
  out[t] = sums[t] / fmaxf(cnt[t >> 7], 1.0f);
}

// =========================================================================
extern "C" void kernel_launch(void* const* d_in, const int* in_sizes, int n_in,
                              void* d_out, int out_size, void* d_ws, size_t ws_size,
                              hipStream_t stream) {
  (void)in_sizes; (void)n_in; (void)out_size; (void)ws_size;
  const float* x     = (const float*)d_in[0];
  const int*   eidx  = (const int*)d_in[1];
  const int*   srcI  = eidx;
  const int*   dstI  = eidx + NEDGES;
  const float* ea    = (const float*)d_in[2];
  const int*   batch = (const int*)d_in[3];
  const float* Wl[3]  = {(const float*)d_in[5],  (const float*)d_in[12], (const float*)d_in[19]};
  const float* bl[3]  = {(const float*)d_in[6],  (const float*)d_in[13], (const float*)d_in[20]};
  const float* Wr[3]  = {(const float*)d_in[7],  (const float*)d_in[14], (const float*)d_in[21]};
  const float* br[3]  = {(const float*)d_in[8],  (const float*)d_in[15], (const float*)d_in[22]};
  const float* We[3]  = {(const float*)d_in[9],  (const float*)d_in[16], (const float*)d_in[23]};
  const float* att[3] = {(const float*)d_in[10], (const float*)d_in[17], (const float*)d_in[24]};
  const float* bo[3]  = {(const float*)d_in[11], (const float*)d_in[18], (const float*)d_in[25]};
  const float* gln[3] = {(const float*)d_in[26], (const float*)d_in[28], (const float*)d_in[30]};
  const float* bln[3] = {(const float*)d_in[27], (const float*)d_in[29], (const float*)d_in[31]};
  const float* Wres   = (const float*)d_in[32];

  // ---- workspace bump allocator ----
  char* ws = (char*)d_ws;
  size_t off = 0;
  auto alloc = [&](size_t bytes) -> void* {
    void* p = ws + off;
    off = (off + bytes + 255) & ~(size_t)255;
    return p;
  };
  __bf16* xbf    = (__bf16*)alloc((size_t)NNODES * NDIM * 2);
  __bf16* hbf    = (__bf16*)alloc((size_t)NNODES * HID * 2);
  __bf16* wl1p   = (__bf16*)alloc((size_t)NDIM * HID * 2);
  __bf16* wr1p   = (__bf16*)alloc((size_t)NDIM * HID * 2);
  __bf16* wresp  = (__bf16*)alloc((size_t)NDIM * HID * 2);
  __bf16* wl2p   = (__bf16*)alloc((size_t)HID * HID * 2);
  __bf16* wr2p   = (__bf16*)alloc((size_t)HID * HID * 2);
  __bf16* wl3p   = (__bf16*)alloc((size_t)HID * OUTD * 2);
  __bf16* wr3p   = (__bf16*)alloc((size_t)HID * OUTD * 2);
  float* xlB     = (float*)alloc((size_t)NNODES * HID * 4);
  float* xrB     = (float*)alloc((size_t)NNODES * HID * 4);
  float* outAcc  = (float*)alloc((size_t)NNODES * HID * 4);
  float* resB    = (float*)alloc((size_t)NNODES * HID * 4);
  float* hsave   = (float*)alloc((size_t)NNODES * HID * 4);
  float* h3      = (float*)alloc((size_t)NNODES * OUTD * 4);
  float* degB    = (float*)alloc((size_t)NNODES * 4);
  float* easumB  = (float*)alloc((size_t)NNODES * 4);
  float* loopEa  = (float*)alloc((size_t)NNODES * 4);
  float* alphaB  = (float*)alloc((size_t)E2TOT * 4 * 4);
  float* exB     = (float*)alloc((size_t)E2TOT * 4 * 4);
  unsigned* amax = (unsigned*)alloc((size_t)NNODES * 4 * 4);
  float* denomB  = (float*)alloc((size_t)NNODES * 4 * 4);
  float* sums    = (float*)alloc((size_t)NGRAPHS * OUTD * 4);
  float* cnt     = (float*)alloc((size_t)NGRAPHS * 4);

  // ---- convert activations; pack weights into WMMA fragment layout ----
  cvt_bf16_kernel<<<(NNODES * NDIM + 255) / 256, 256, 0, stream>>>(x, xbf, NNODES * NDIM);
  auto pack = [&](const float* B, __bf16* P, int K, int N) {
    pack_b_kernel<<<(K * N + 255) / 256, 256, 0, stream>>>(B, P, K, N);
  };
  pack(Wl[0], wl1p, NDIM, HID);  pack(Wr[0], wr1p, NDIM, HID);
  pack(Wres,  wresp, NDIM, HID);
  pack(Wl[1], wl2p, HID, HID);   pack(Wr[1], wr2p, HID, HID);
  pack(Wl[2], wl3p, HID, OUTD);  pack(Wr[2], wr3p, HID, OUTD);

  // ---- self-loop edge attr (once; graph fixed across layers) ----
  hipMemsetAsync(degB, 0, NNODES * 4, stream);
  hipMemsetAsync(easumB, 0, NNODES * 4, stream);
  deg_easum_kernel<<<(NEDGES + 255) / 256, 256, 0, stream>>>(dstI, ea, degB, easumB, NEDGES);
  loopea_kernel<<<(NNODES + 255) / 256, 256, 0, stream>>>(easumB, degB, loopEa, NNODES);

  const int edgeWaveBlocks = E2TOT / 8;   // one wave per edge, 8 waves/block

  for (int l = 0; l < 3; ++l) {
    const __bf16* Xb   = (l == 0) ? xbf : hbf;
    const int K        = (l == 0) ? NDIM : HID;
    const int dout     = (l == 2) ? OUTD : HID;
    const int H        = (l == 2) ? 1 : 4;
    const __bf16* wlb  = (l == 0) ? wl1p : (l == 1) ? wl2p : wl3p;
    const __bf16* wrb  = (l == 0) ? wr1p : (l == 1) ? wr2p : wr3p;

    dim3 gg(dout / 64, NNODES / 32);
    gemm_bf16_wmma<<<gg, 32, 0, stream>>>(Xb, wlb, bl[l], xlB, NNODES, K, dout);
    gemm_bf16_wmma<<<gg, 32, 0, stream>>>(Xb, wrb, br[l], xrB, NNODES, K, dout);
    if (l == 0)
      gemm_bf16_wmma<<<gg, 32, 0, stream>>>(xbf, wresp, nullptr, resB, NNODES, NDIM, HID);

    hipMemsetAsync(amax, 0, (size_t)NNODES * H * 4, stream);   // key 0 == -inf
    hipMemsetAsync(denomB, 0, (size_t)NNODES * H * 4, stream);
    hipMemsetAsync(outAcc, 0, (size_t)NNODES * dout * 4, stream);

    if (l < 2) {
      edge_alpha_kernel<HID, 4><<<edgeWaveBlocks, 256, 0, stream>>>(
          srcI, dstI, ea, loopEa, xlB, xrB, We[l], att[l], alphaB, amax);
      edge_exp_kernel<4><<<(E2TOT * 4 + 255) / 256, 256, 0, stream>>>(
          dstI, alphaB, amax, exB, denomB);
      edge_scatter_kernel<HID, 4><<<edgeWaveBlocks, 256, 0, stream>>>(
          srcI, dstI, xlB, exB, denomB, outAcc);
    } else {
      edge_alpha_kernel<OUTD, 1><<<edgeWaveBlocks, 256, 0, stream>>>(
          srcI, dstI, ea, loopEa, xlB, xrB, We[l], att[l], alphaB, amax);
      edge_exp_kernel<1><<<(E2TOT + 255) / 256, 256, 0, stream>>>(
          dstI, alphaB, amax, exB, denomB);
      edge_scatter_kernel<OUTD, 1><<<edgeWaveBlocks, 256, 0, stream>>>(
          srcI, dstI, xlB, exB, denomB, outAcc);
    }

    if (l == 0)
      node_post_kernel<HID><<<NNODES, 32, 0, stream>>>(outAcc, bo[0], gln[0], bln[0],
                                                       resB, hsave, hbf, 1);
    else if (l == 1)
      node_post_kernel<HID><<<NNODES, 32, 0, stream>>>(outAcc, bo[1], gln[1], bln[1],
                                                       hsave, hsave, hbf, 1);
    else
      node_post_kernel<OUTD><<<NNODES, 32, 0, stream>>>(outAcc, bo[2], gln[2], bln[2],
                                                        nullptr, h3, nullptr, 0);
  }

  // ---- global mean pool ----
  hipMemsetAsync(sums, 0, (size_t)NGRAPHS * OUTD * 4, stream);
  hipMemsetAsync(cnt, 0, (size_t)NGRAPHS * 4, stream);
  pool_scatter_kernel<<<(NNODES * OUTD + 255) / 256, 256, 0, stream>>>(h3, batch, sums, cnt);
  pool_div_kernel<<<(NGRAPHS * OUTD + 255) / 256, 256, 0, stream>>>(sums, cnt, (float*)d_out);
}
